// GAT_42176578847122
// MI455X (gfx1250) — matrix operand order repaired and checked by reading, not compiled, not run on hardware
//
#include <hip/hip_runtime.h>

// GAT attention, flash-decoding style, CDNA5 (gfx1250, wave32, WMMA f16->f32).
// B=4, N=4096, D=128. adj (256 MB int32) is the HBM floor (~11us @ 23.3 TB/s).
// 4 waves split the key dim per 16-query tile (4096 waves total for latency
// hiding), 64 keys per inner block, softmax kept in exp2 domain with the
// 1/sqrt(D)*log2(e) scale pre-folded into the Q fragments.

#define BATCH 4
#define NKEYS 4096
#define DIM   128
#define KSPLIT 4
#define KRANGE (NKEYS / KSPLIT)
#define C2 0.12753102f /* log2(e)/sqrt(128) */

typedef __attribute__((ext_vector_type(16))) _Float16 v16h;
typedef __attribute__((ext_vector_type(8)))  _Float16 v8h;
typedef __attribute__((ext_vector_type(8)))  float    v8f;

// ---------------------------------------------------------------------------
// Pass 1: fp32 X -> f16 row-major (Xh[b][n][d]) AND f16 transposed
// (Xt[b][d][n]) via an LDS-tiled transpose; both global streams coalesced.
// ---------------------------------------------------------------------------
__global__ __launch_bounds__(256) void gat_cvt_transpose(
    const float* __restrict__ X, _Float16* __restrict__ Xh,
    _Float16* __restrict__ Xt)
{
  __shared__ _Float16 tile[32][140];
  const int b  = blockIdx.y;
  const int n0 = blockIdx.x * 32;
  const int t  = threadIdx.x;
  const float* src  = X  + ((size_t)b * NKEYS + n0) * DIM;
  _Float16*    dstH = Xh + ((size_t)b * NKEYS + n0) * DIM;
#pragma unroll
  for (int i = 0; i < 16; ++i) {
    int e  = t + i * 256;
    int nn = e >> 7, d = e & 127;
    _Float16 h = (_Float16)src[e];
    dstH[e]     = h;
    tile[nn][d] = h;
  }
  __syncthreads();
#pragma unroll
  for (int i = 0; i < 16; ++i) {
    int e  = t + i * 256;
    int d  = e >> 5, nn = e & 31;
    Xt[((size_t)b * DIM + d) * NKEYS + n0 + nn] = tile[nn][d];
  }
}

// ---------------------------------------------------------------------------
// Pass 2: fused masked attention. Block = 128 threads = 4 waves = 1 query
// tile; wave w owns keys [w*1024, (w+1)*1024), results merged through LDS.
// C/D layout: lane L -> col n = L&15, VGPR r -> row m = (L>>4)*8 + r.
// A layout:   lane L -> row m = L&15, halves: k = (h<8?0:16) + (L>>4)*8 + (h&7).
// B layout:   lane L -> col n = L&15, halves: k = (h<8?0:16) + (L>>4)*8 + (h&7), same as A.
// ---------------------------------------------------------------------------
__global__ __launch_bounds__(128) void gat_flash_attn(
    const _Float16* __restrict__ Xh, const _Float16* __restrict__ Xt,
    const int* __restrict__ adj, float* __restrict__ out)
{
  // P staging: 16 rows x 64 keys, row stride 72 halves -> conflict-free b128
  // A-frag reads (m*36 mod 64 distinct for m=0..15).
  __shared__ __align__(32) _Float16 pbuf[4][16 * 72];
  __shared__ __align__(16) float obuf[KSPLIT - 1][16][DIM]; // partial O, waves 1..3
  __shared__ float mstat[KSPLIT - 1][16];
  __shared__ float lstat[KSPLIT - 1][16];

  const int w    = threadIdx.x >> 5;
  const int lane = threadIdx.x & 31;
  const int nn   = lane & 15;
  const int hi   = lane >> 4;

  const int tileId = blockIdx.x;              // 0..1023
  const int b      = tileId >> 8;
  const int q0     = (tileId & 255) << 4;

  // ---- Q A-fragments, pre-scaled into exp2 score domain ----
  v16h qf[4];
  {
    const _Float16* qrow = Xh + ((size_t)b * NKEYS + q0 + nn) * DIM;
#pragma unroll
    for (int c = 0; c < 4; ++c) {
      v8h lo = *(const v8h*)(qrow + c * 32 + hi * 8);
      v8h h8 = *(const v8h*)(qrow + c * 32 + 16 + hi * 8);
      v16h q = __builtin_shufflevector(lo, h8,
               0,1,2,3,4,5,6,7,8,9,10,11,12,13,14,15);
      qf[c] = q * (_Float16)C2;
    }
  }

  v8f acc[8];
#pragma unroll
  for (int c = 0; c < 8; ++c) acc[c] = (v8f){};
  float mrow[8], lrow[8];
#pragma unroll
  for (int r = 0; r < 8; ++r) { mrow[r] = -__builtin_huge_valf(); lrow[r] = 0.0f; }

  const _Float16* kbase   = Xh + (size_t)b * NKEYS * DIM;
  const _Float16* vbase   = Xt + (size_t)b * DIM * NKEYS;
  const int*      adjbase = adj + ((size_t)b * NKEYS + q0 + hi * 8) * (size_t)NKEYS + nn;

  const int k0 = w * KRANGE;
  for (int kb = k0; kb < k0 + KRANGE; kb += 64) {
    __builtin_prefetch(adjbase + kb + 64, 0, 1);   // global_prefetch_b8

    // ---- S = Q*K^T for 64 keys: four 16x16 C tiles (exp2 domain) ----
    v8f st[4];
#pragma unroll
    for (int t = 0; t < 4; ++t) {
      const _Float16* krow = kbase + (size_t)(kb + t * 16 + nn) * DIM;
      v8f s = (v8f){};
#pragma unroll
      for (int c = 0; c < 4; ++c) {
        v8h klo = *(const v8h*)(krow + c * 32 + hi * 8);
        v8h khi = *(const v8h*)(krow + c * 32 + 16 + hi * 8);
        v16h kf = __builtin_shufflevector(klo, khi, 0,1,2,3,4,5,6,7,8,9,10,11,12,13,14,15);
        s = __builtin_amdgcn_wmma_f32_16x16x32_f16(false, qf[c], false, kf,
                                                   (short)0, s, false, false);
      }
      st[t] = s;
    }

    // ---- mask + joint online softmax over the 64-key block ----
    const int* arow = adjbase + kb;
    float corr[8];
#pragma unroll
    for (int r = 0; r < 8; ++r) {
      float sv[4];
#pragma unroll
      for (int t = 0; t < 4; ++t) {
        int a = arow[(size_t)r * NKEYS + t * 16];
        sv[t] = (a > 0) ? st[t][r] : -1.0e12f;
      }
      float rm = fmaxf(fmaxf(sv[0], sv[1]), fmaxf(sv[2], sv[3]));
      rm = fmaxf(rm, __shfl_xor(rm, 1, 32));
      rm = fmaxf(rm, __shfl_xor(rm, 2, 32));
      rm = fmaxf(rm, __shfl_xor(rm, 4, 32));
      rm = fmaxf(rm, __shfl_xor(rm, 8, 32));
      float newm = fmaxf(mrow[r], rm);
      corr[r] = exp2f(mrow[r] - newm);
      float ps = 0.0f;
#pragma unroll
      for (int t = 0; t < 4; ++t) {
        float p = exp2f(sv[t] - newm);
        ps += p;
        pbuf[w][(hi * 8 + r) * 72 + t * 16 + nn] = (_Float16)p;
      }
      ps += __shfl_xor(ps, 1, 32);
      ps += __shfl_xor(ps, 2, 32);
      ps += __shfl_xor(ps, 4, 32);
      ps += __shfl_xor(ps, 8, 32);
      lrow[r] = lrow[r] * corr[r] + ps;
      mrow[r] = newm;
    }
#pragma unroll
    for (int c = 0; c < 8; ++c)
#pragma unroll
      for (int r = 0; r < 8; ++r) acc[c][r] *= corr[r];

    // ---- P A-fragments (two 16x32) from LDS; same-wave DS is in-order ----
    v16h pf[2];
#pragma unroll
    for (int s2 = 0; s2 < 2; ++s2) {
      v8h plo = *(const v8h*)(&pbuf[w][nn * 72 + s2 * 32 + hi * 8]);
      v8h phi = *(const v8h*)(&pbuf[w][nn * 72 + s2 * 32 + 16 + hi * 8]);
      pf[s2] = __builtin_shufflevector(plo, phi,
               0,1,2,3,4,5,6,7,8,9,10,11,12,13,14,15);
    }

    // ---- O += P*V, V B-frags contiguous from transposed X ----
#pragma unroll
    for (int c = 0; c < 8; ++c) {
      const _Float16* vrow = vbase + (size_t)(c * 16 + nn) * NKEYS + kb;
#pragma unroll
      for (int s2 = 0; s2 < 2; ++s2) {
        v8h vlo = *(const v8h*)(vrow + s2 * 32 + hi * 8);
        v8h vhi = *(const v8h*)(vrow + s2 * 32 + 16 + hi * 8);
        v16h vf = __builtin_shufflevector(vlo, vhi, 0,1,2,3,4,5,6,7,8,9,10,11,12,13,14,15);
        acc[c] = __builtin_amdgcn_wmma_f32_16x16x32_f16(false, pf[s2], false, vf,
                                                        (short)0, acc[c], false, false);
      }
    }
  }

  // ---- cross-wave merge of the 4 key partitions ----
  if (w > 0) {
    if (nn == 0) {
#pragma unroll
      for (int r = 0; r < 8; ++r) {
        mstat[w - 1][hi * 8 + r] = mrow[r];
        lstat[w - 1][hi * 8 + r] = lrow[r];
      }
    }
#pragma unroll
    for (int r = 0; r < 8; ++r)
#pragma unroll
      for (int c = 0; c < 8; ++c)
        obuf[w - 1][hi * 8 + r][c * 16 + nn] = acc[c][r];
  }
  __syncthreads();

  if (w == 0) {
    float* orow = out + ((size_t)b * NKEYS + q0 + hi * 8) * DIM;
#pragma unroll
    for (int r = 0; r < 8; ++r) {
      const int m = hi * 8 + r;
      float M = mrow[r];
#pragma unroll
      for (int j = 0; j < KSPLIT - 1; ++j) M = fmaxf(M, mstat[j][m]);
      float f0 = exp2f(mrow[r] - M);
      float l  = lrow[r] * f0;
      float fj[KSPLIT - 1];
#pragma unroll
      for (int j = 0; j < KSPLIT - 1; ++j) {
        fj[j] = exp2f(mstat[j][m] - M);
        l += lstat[j][m] * fj[j];
      }
      float inv = 1.0f / l;
#pragma unroll
      for (int c = 0; c < 8; ++c) {
        float o = acc[c][r] * f0;
#pragma unroll
        for (int j = 0; j < KSPLIT - 1; ++j)
          o += obuf[j][m][c * 16 + nn] * fj[j];
        orow[(size_t)r * DIM + c * 16 + nn] = o * inv;
      }
    }
  }
}

// ---------------------------------------------------------------------------
extern "C" void kernel_launch(void* const* d_in, const int* in_sizes, int n_in,
                              void* d_out, int out_size, void* d_ws, size_t ws_size,
                              hipStream_t stream) {
  const float* X   = (const float*)d_in[0];
  const int*   adj = (const int*)d_in[1];
  float*       out = (float*)d_out;

  // workspace: Xh (4 MB f16 row-major) + Xt (4 MB f16 transposed) = 8 MB
  _Float16* Xh = (_Float16*)d_ws;
  _Float16* Xt = Xh + (size_t)BATCH * NKEYS * DIM;

  gat_cvt_transpose<<<dim3(NKEYS / 32, BATCH), 256, 0, stream>>>(X, Xh, Xt);
  gat_flash_attn<<<dim3(BATCH * NKEYS / 16), 128, 0, stream>>>(Xh, Xt, adj, out);
}
